// FlamingoPaLM_5669356835462
// MI455X (gfx1250) — matrix operand order repaired and checked by reading, not compiled, not run on hardware
//
// FlamingoPaLM forward for MI455X (gfx1250, wave32, WMMA).
// All GEMMs route through v_wmma_f32_16x16x32_bf16 (bf16 inputs, f32 accum).
// One batched/strided GEMM kernel covers every einsum (head batching, MQA
// shared K/V via zero strides, transposed-B for q@k^T and tied-embedding
// logits, ldc/strideC tricks to fuse the [H,N,DH]->[N,INNER] merge).
//
// GEMM data path:
//   - staging split into load-phase (8x global_load_b128 clause into regs)
//     then convert/store-phase (fp32->bf16 RNE, packed ushort4 LDS stores),
//     so HBM latency is exposed once per round, not per load.
//   - double-buffered LDS tiles (ping-pong): ONE barrier per K-round.
//   - A staged [m][k]; B staged TRANSPOSED [n][k] so a lane's fragment is
//     16 contiguous bf16 -> fragments are built from ds_load_b128 only.
//   - rows padded to 72 ushorts (144B: 16B chunks aligned, stride 36 banks).
//   - BK=64: 4 WMMAs per wave per barrier round (all K here are % 64 == 0).
//
// Input order assumption: harness flattens setup_inputs() like a JAX pytree
// (dict keys sorted alphabetically, None subtrees dropped):
//   embeds,
//   params.final_g,
//   params.layers[i]: (gca: a_b,a_g,attn_gate,ff_b,ff_g,ff_gate,ff_w1,ff_w2,
//                            wkv,wo,wq   -- only when i%3==0)
//                     (ptb: attn_out, ff_out, fused, g),
//   params.perc: latents,
//                layers[j]: ff_b,ff_g,ff_w1,ff_w2,nl_b,nl_g,nm_b,nm_g,wkv,wo,wq,
//                media_pos, norm_b, norm_g,
//   params.token_emb,
//   text
//
// Shape facts used (no edge handling needed): every GEMM M % 64 == 0,
// N % 64 == 0, K % 64 == 0 for this model; all pointers/lds/strides are
// multiples of 4 floats (16B vector alignment holds).

#include <hip/hip_runtime.h>
#include <hip/hip_bf16.h>
#include <math.h>

namespace cfg {
constexpr int D = 1024, H = 16, DH = 64, V = 32000, DEPTH = 8, CROSS_EVERY = 3;
constexpr int FF_IN = 4096, P_DEPTH = 2, N_LAT = 64, T = 4, NPATCH = 256, NSEQ = 1024;
constexpr int INNER = H * DH;                       // 1024
constexpr int FUSED_W = INNER + DH + DH + 2 * FF_IN; // 9344
constexpr float SCALE = 0.125f;                      // DH^-0.5
constexpr int K_OFF = INNER;                         // 1024
constexpr int V_OFF = INNER + DH;                    // 1088
constexpr int XG_OFF = INNER + 2 * DH;               // 1152
constexpr int GATE_OFF = XG_OFF + FF_IN;             // 5248
}

// ----------------------------- WMMA GEMM ------------------------------------

typedef __attribute__((ext_vector_type(16))) __bf16          v16bf;
typedef __attribute__((ext_vector_type(8)))  __bf16          v8bf;
typedef __attribute__((ext_vector_type(8)))  float           v8f;
typedef __attribute__((ext_vector_type(4)))  float           f4;
typedef __attribute__((ext_vector_type(4)))  unsigned short  us4;

__device__ __forceinline__ unsigned short f2bf_bits(float f) {
  union { float f; unsigned u; } cv; cv.f = f;
  // round-to-nearest-even fp32 -> bf16
  return (unsigned short)((cv.u + 0x7fffu + ((cv.u >> 16) & 1u)) >> 16);
}
__device__ __forceinline__ us4 f4_to_bf4(f4 v) {
  us4 r;
  r[0] = f2bf_bits(v[0]); r[1] = f2bf_bits(v[1]);
  r[2] = f2bf_bits(v[2]); r[3] = f2bf_bits(v[3]);
  return r;
}
__device__ __forceinline__ v16bf cat8(v8bf lo, v8bf hi) {
  return __builtin_shufflevector(lo, hi, 0, 1, 2, 3, 4, 5, 6, 7,
                                 8, 9, 10, 11, 12, 13, 14, 15);
}

// C[z] = alpha * A[z] @ (TRANSB ? B[z]^T : B[z])  (+ C[z] if accumulate)
// Block tile 64(M) x 64(N) x 64(K). 256 threads = 8 waves; wave (wm,wn) owns a
// 16x32 strip -> 2 accumulators, 4 v_wmma_f32_16x16x32_bf16 per K-round.
template <bool TRANSB>
__global__ __launch_bounds__(256) void gemm_wmma_bf16(
    const float* __restrict__ A, const float* __restrict__ B, float* __restrict__ C,
    int M, int N, int K, int lda, int ldb, int ldc,
    long long sA1, long long sA2, long long sB1, long long sB2,
    long long sC1, long long sC2, int nb2,
    float alpha, const float* __restrict__ alphaDev, int accumulate) {
  // Ping-pong tiles. Rows padded to 72 ushorts = 144B: 16B chunks stay
  // aligned, row stride covers 36 banks (scatters stores + b128 reads).
  __shared__ unsigned short As[2][64][72];  // [m][k] bf16 bits
  __shared__ unsigned short Bs[2][64][72];  // [n][k] bf16 bits (transposed)

  const int tid  = threadIdx.x;
  const int lane = tid & 31;
  const int wave = tid >> 5;
  const int wm   = wave & 3;   // M sub-tile 0..3
  const int wn   = wave >> 2;  // N group of 32: 0..1
  const int z    = blockIdx.z;
  const int z1   = z / nb2, z2 = z - z1 * nb2;
  A += z1 * sA1 + z2 * sA2;
  B += z1 * sB1 + z2 * sB2;
  C += z1 * sC1 + z2 * sC2;
  const int row0 = blockIdx.y * 64;
  const int col0 = blockIdx.x * 64;

  const int half = lane >> 4;   // 0 or 1
  const int r16  = lane & 15;

  v8f acc0 = {}; v8f acc1 = {};
  int buf = 0;

  for (int k0 = 0; k0 < K; k0 += 64, buf ^= 1) {
    // ---- load phase: issue all 8 b128 loads before any conversion ----
    f4 va[4], vb[4];
#pragma unroll
    for (int e = 0; e < 4; ++e) {
      int i = e * 256 + tid;        // 0..1023 float4 slots
      va[e] = *(const f4*)&A[(size_t)(row0 + (i >> 4)) * lda + (k0 + ((i & 15) << 2))];
    }
    if (TRANSB) {
#pragma unroll
      for (int e = 0; e < 4; ++e) {
        int i = e * 256 + tid;
        vb[e] = *(const f4*)&B[(size_t)(col0 + (i >> 4)) * ldb + (k0 + ((i & 15) << 2))];
      }
    } else {
#pragma unroll
      for (int e = 0; e < 4; ++e) {
        int i = e * 256 + tid;
        vb[e] = *(const f4*)&B[(size_t)(k0 + (i >> 4)) * ldb + (col0 + ((i & 15) << 2))];
      }
    }
    // prefetch next K tile (global_prefetch_b8)
    if (k0 + 64 < K) {
      __builtin_prefetch(&A[(size_t)(row0 + (tid >> 6)) * lda + (k0 + 64 + (tid & 63))], 0, 1);
      if (TRANSB)
        __builtin_prefetch(&B[(size_t)(col0 + (tid >> 6)) * ldb + (k0 + 64 + (tid & 63))], 0, 1);
      else
        __builtin_prefetch(&B[(size_t)(k0 + 64 + (tid >> 6)) * ldb + (col0 + (tid & 63))], 0, 1);
    }

    // ---- convert/store phase: packed ushort4 LDS stores ----
#pragma unroll
    for (int e = 0; e < 4; ++e) {
      int i = e * 256 + tid;
      *(us4*)&As[buf][i >> 4][(i & 15) << 2] = f4_to_bf4(va[e]);
    }
    if (TRANSB) {
#pragma unroll
      for (int e = 0; e < 4; ++e) {
        int i = e * 256 + tid;
        *(us4*)&Bs[buf][i >> 4][(i & 15) << 2] = f4_to_bf4(vb[e]);
      }
    } else {
#pragma unroll
      for (int e = 0; e < 4; ++e) {
        int i = e * 256 + tid;
        int kk = i >> 4;            // k 0..63
        int n  = (i & 15) << 2;     // n 0..60 step 4
        us4 p = f4_to_bf4(vb[e]);
        Bs[buf][n + 0][kk] = p[0];
        Bs[buf][n + 1][kk] = p[1];
        Bs[buf][n + 2][kk] = p[2];
        Bs[buf][n + 3][kk] = p[3];
      }
    }
    // Single barrier per round (double buffering makes the trailing WAR
    // barrier unnecessary: next round stores to the other buffer, and the
    // round after that is ordered behind the next round's barrier).
    __syncthreads();

    // ---- fragments via b128 LDS loads only ----
    // A 16x32 layout: lanes 0-15 row=lane, K 0..15; lanes 16-31
    //   row=lane-16, K 16..31  -> chunks (kk*4+2*half, kk*4+2*half+1).
    // B 32x16 layout: lanes 0-15 col=lane K 0..15; lanes 16-31 col=lane-16
    //   K 16..31 -> contiguous 16 bf16 at k = kk*32 + half*16.
    const v8bf* arow  = (const v8bf*)(&As[buf][wm * 16 + r16][0]);
    const v8bf* brow0 = (const v8bf*)(&Bs[buf][wn * 32 + r16][0]);
    const v8bf* brow1 = (const v8bf*)(&Bs[buf][wn * 32 + 16 + r16][0]);
#pragma unroll
    for (int kk = 0; kk < 2; ++kk) {
      v16bf a  = cat8(arow[kk * 4 + 2 * half], arow[kk * 4 + 2 * half + 1]);
      v16bf b0 = cat8(brow0[kk * 4 + 2 * half], brow0[kk * 4 + 2 * half + 1]);
      v16bf b1 = cat8(brow1[kk * 4 + 2 * half], brow1[kk * 4 + 2 * half + 1]);
      acc0 = __builtin_amdgcn_wmma_f32_16x16x32_bf16(false, a, false, b0, (short)0,
                                                     acc0, false, false);
      acc1 = __builtin_amdgcn_wmma_f32_16x16x32_bf16(false, a, false, b1, (short)0,
                                                     acc1, false, false);
    }
  }

  float am = alpha;
  if (alphaDev) am *= tanhf(alphaDev[0]);
  // C/D 16x16 f32: lanes 0-15 VGPR r -> M=r, N=lane; lanes 16-31 M=8+r.
  const int mBase = row0 + wm * 16 + half * 8;
  const int nB    = col0 + wn * 32 + r16;
#pragma unroll
  for (int r = 0; r < 8; ++r) {
    size_t o0 = (size_t)(mBase + r) * ldc + nB;
    size_t o1 = o0 + 16;
    float v0 = am * acc0[r], v1 = am * acc1[r];
    if (accumulate) { v0 += C[o0]; v1 += C[o1]; }
    C[o0] = v0; C[o1] = v1;
  }
}

// ----------------------------- support kernels ------------------------------

__global__ __launch_bounds__(256) void layernorm_kernel(
    const float* __restrict__ in, const float* __restrict__ g,
    const float* __restrict__ b, float* __restrict__ out, int D) {
  const int row = blockIdx.x, tid = threadIdx.x;
  const float* xr = in + (size_t)row * D;
  __shared__ float red[256];
  float s = 0.f;
  for (int j = tid; j < D; j += 256) s += xr[j];
  red[tid] = s; __syncthreads();
  for (int k = 128; k > 0; k >>= 1) { if (tid < k) red[tid] += red[tid + k]; __syncthreads(); }
  float mean = red[0] / D; __syncthreads();
  float v = 0.f;
  for (int j = tid; j < D; j += 256) { float d = xr[j] - mean; v += d * d; }
  red[tid] = v; __syncthreads();
  for (int k = 128; k > 0; k >>= 1) { if (tid < k) red[tid] += red[tid + k]; __syncthreads(); }
  float inv = rsqrtf(red[0] / D + 1e-5f);
  for (int j = tid; j < D; j += 256)
    out[(size_t)row * D + j] = (xr[j] - mean) * inv * g[j] + (b ? b[j] : 0.0f);
}

// mode 0 plain; 1 causal (col > row_in_seq masked); 2 GCA media-time mask.
__global__ __launch_bounds__(256) void softmax_kernel(
    float* __restrict__ S, int cols, int seqN, int mode,
    const int* __restrict__ ttime, int mdiv) {
  const int r = blockIdx.x, tid = threadIdx.x;
  float* row = S + (size_t)r * cols;
  const int n  = r % seqN;
  const int tt = (mode == 2) ? ttime[n] : 0;
  __shared__ float red[256];
  float vloc[4];
  const float NEG = -3.402823466e38f;
  int cnt = 0; float mx = NEG;
  for (int j = tid; j < cols; j += 256) {
    float v = row[j];
    if (mode == 1 && j > n) v = NEG;
    if (mode == 2 && tt != 1 + j / mdiv) v = NEG;
    vloc[cnt++] = v;
    mx = fmaxf(mx, v);
  }
  red[tid] = mx; __syncthreads();
  for (int k = 128; k > 0; k >>= 1) { if (tid < k) red[tid] = fmaxf(red[tid], red[tid + k]); __syncthreads(); }
  mx = red[0]; __syncthreads();
  float sum = 0.f; cnt = 0;
  for (int j = tid; j < cols; j += 256) {
    float e = expf(vloc[cnt] - mx);
    vloc[cnt++] = e; sum += e;
  }
  red[tid] = sum; __syncthreads();
  for (int k = 128; k > 0; k >>= 1) { if (tid < k) red[tid] += red[tid + k]; __syncthreads(); }
  float inv = 1.0f / red[0];
  int zero = (mode == 2 && tt == 0);
  cnt = 0;
  for (int j = tid; j < cols; j += 256)
    row[j] = zero ? 0.0f : vloc[cnt++] * inv;
}

__global__ void gather_emb_kernel(const float* __restrict__ emb,
                                  const int* __restrict__ text,
                                  float* __restrict__ out, int D) {
  int n = blockIdx.x;
  int t = text[n];
  for (int j = threadIdx.x; j < D; j += blockDim.x)
    out[(size_t)n * D + j] = emb[(size_t)t * D + j];
}

__global__ void cumsum_media_kernel(const int* __restrict__ text,
                                    int* __restrict__ tt, int n) {
  __shared__ int buf[1024];
  int i = threadIdx.x;
  buf[i] = (text[i] == 3) ? 1 : 0;  // MEDIA_ID == 3
  __syncthreads();
  for (int off = 1; off < n; off <<= 1) {
    int v = (i >= off) ? buf[i - off] : 0;
    __syncthreads();
    buf[i] += v;
    __syncthreads();
  }
  tt[i] = buf[i];
}

// RoPE gather from the fused projection: out[(h*seqN+n)*64+d]
__global__ void rope_kernel(const float* __restrict__ in, float* __restrict__ out,
                            int seqN, int ldin, int coloff) {
  int n = blockIdx.x, h = blockIdx.y, d = threadIdx.x;  // block of 64
  const float* src = in + (size_t)n * ldin + coloff + h * 64;
  int i = d & 31;
  float p   = (float)n * powf(10000.0f, -(float)(2 * i) * (1.0f / 64.0f));
  float t   = src[d];
  float rot = (d < 32) ? -src[d + 32] : src[d - 32];
  out[((size_t)h * seqN + n) * 64 + d] = t * cosf(p) + rot * sinf(p);
}

__global__ void gate_silu_kernel(const float* __restrict__ proj,
                                 float* __restrict__ out, int count) {
  size_t i = (size_t)blockIdx.x * blockDim.x + threadIdx.x;
  if (i >= (size_t)count) return;
  int n = (int)(i >> 12);       // / 4096
  int j = (int)(i & 4095);
  float xg = proj[(size_t)n * cfg::FUSED_W + cfg::XG_OFF + j];
  float gt = proj[(size_t)n * cfg::FUSED_W + cfg::GATE_OFF + j];
  out[i] = (gt / (1.0f + expf(-gt))) * xg;  // silu(gate) * xg
}

__global__ void gelu_kernel(float* __restrict__ v, int count) {
  size_t i = (size_t)blockIdx.x * blockDim.x + threadIdx.x;
  if (i >= (size_t)count) return;
  float t = v[i];
  v[i] = 0.5f * t * (1.0f + erff(t * 0.70710678118654752f));  // exact gelu
}

__global__ void media_add_kernel(const float* __restrict__ embeds,
                                 const float* __restrict__ mpos,
                                 float* __restrict__ out, int count) {
  size_t i = (size_t)blockIdx.x * blockDim.x + threadIdx.x;
  if (i >= (size_t)count) return;
  int t = (int)(i >> 18);   // / (256*1024)
  int d = (int)(i & 1023);
  out[i] = embeds[i] + mpos[t * 1024 + d];  // media_pos[t,0,:]
}

__global__ void lat_init_kernel(const float* __restrict__ latents,
                                float* __restrict__ lat, int count) {
  size_t i = (size_t)blockIdx.x * blockDim.x + threadIdx.x;
  if (i >= (size_t)count) return;
  lat[i] = latents[i & 65535];  // broadcast [64,1024] over t
}

__global__ void kvin_kernel(const float* __restrict__ xm,
                            const float* __restrict__ lm,
                            float* __restrict__ kvin, int count) {
  size_t i = (size_t)blockIdx.x * blockDim.x + threadIdx.x;
  if (i >= (size_t)count) return;
  int d = (int)(i & 1023);
  int r = (int)((i >> 10) % 320);
  int t = (int)(i / (size_t)(320 * 1024));
  kvin[i] = (r < 256) ? xm[((size_t)(t * 256 + r)) * 1024 + d]
                      : lm[((size_t)(t * 64 + (r - 256))) * 1024 + d];
}

// ----------------------------- host orchestration ---------------------------

static void gemm(hipStream_t s, const float* A, const float* B, float* C,
                 int M, int N, int K, int lda, int ldb, int ldc,
                 long long sA1, long long sA2, long long sB1, long long sB2,
                 long long sC1, long long sC2, int b1, int b2,
                 float alpha, const float* aDev, int acc, bool transB) {
  dim3 grid(N / 64, M / 64, b1 * b2), block(256);
  if (transB)
    gemm_wmma_bf16<true><<<grid, block, 0, s>>>(A, B, C, M, N, K, lda, ldb, ldc,
        sA1, sA2, sB1, sB2, sC1, sC2, b2, alpha, aDev, acc);
  else
    gemm_wmma_bf16<false><<<grid, block, 0, s>>>(A, B, C, M, N, K, lda, ldb, ldc,
        sA1, sA2, sB1, sB2, sC1, sC2, b2, alpha, aDev, acc);
}
static void gemm0(hipStream_t s, const float* A, const float* B, float* C,
                  int M, int N, int K, int lda, int ldb, int ldc,
                  float alpha, const float* aDev, int acc, bool transB) {
  gemm(s, A, B, C, M, N, K, lda, ldb, ldc, 0, 0, 0, 0, 0, 0, 1, 1,
       alpha, aDev, acc, transB);
}

extern "C" void kernel_launch(void* const* d_in, const int* in_sizes, int n_in,
                              void* d_out, int out_size, void* d_ws, size_t ws_size,
                              hipStream_t stream) {
  using namespace cfg;
  (void)in_sizes; (void)n_in; (void)out_size; (void)ws_size;

  int idx = 0;
  auto F = [&](void) { return (const float*)d_in[idx++]; };

  const float* embeds  = F();   // [4,256,1024]
  const float* final_g = F();   // [1024]

  struct GcaP { const float *a_b,*a_g,*attn_gate,*ff_b,*ff_g,*ff_gate,
                            *ff_w1,*ff_w2,*wkv,*wo,*wq; };
  struct PtbP { const float *attn_out,*ff_out,*fused,*g; };
  GcaP gca[DEPTH]; PtbP ptb[DEPTH]; bool hasG[DEPTH];
  for (int i = 0; i < DEPTH; ++i) {
    hasG[i] = (i % CROSS_EVERY == 0);
    if (hasG[i]) {
      gca[i].a_b = F();  gca[i].a_g = F();  gca[i].attn_gate = F();
      gca[i].ff_b = F(); gca[i].ff_g = F(); gca[i].ff_gate = F();
      gca[i].ff_w1 = F(); gca[i].ff_w2 = F(); gca[i].wkv = F();
      gca[i].wo = F();   gca[i].wq = F();
    }
    ptb[i].attn_out = F(); ptb[i].ff_out = F(); ptb[i].fused = F(); ptb[i].g = F();
  }
  const float* latents = F();   // [64,1024]
  struct PercL { const float *ff_b,*ff_g,*ff_w1,*ff_w2,*nl_b,*nl_g,*nm_b,*nm_g,
                             *wkv,*wo,*wq; };
  PercL pl[P_DEPTH];
  for (int j = 0; j < P_DEPTH; ++j) {
    pl[j].ff_b = F(); pl[j].ff_g = F(); pl[j].ff_w1 = F(); pl[j].ff_w2 = F();
    pl[j].nl_b = F(); pl[j].nl_g = F(); pl[j].nm_b = F(); pl[j].nm_g = F();
    pl[j].wkv = F();  pl[j].wo = F();   pl[j].wq = F();
  }
  const float* media_pos = F();  // [4,1,1024]
  const float* norm_b    = F();
  const float* norm_g    = F();
  const float* token_emb = F();  // [32000,1024]
  const int*   text      = (const int*)d_in[idx++];  // [1,1024]

  // --------------- workspace arena (floats); ~152 MB total ----------------
  float* W = (float*)d_ws;
  size_t off = 0;
  auto alloc = [&](size_t n) { float* p = W + off; off += n; return p; };
  const size_t M1 = (size_t)1024 * 1024;
  float* x    = alloc(M1);                    // residual stream [1024,1024]
  float* med  = alloc((size_t)256 * 1024);    // perceiver output [256,1024]
  int*   ttim = (int*)alloc(1024);            // text_time
  float* xn   = alloc(M1);                    // LN scratch (reused)
  float* proj = alloc((size_t)1024 * FUSED_W);// fused proj / kv buffers (reused)
  float* qr   = alloc(M1);                    // q (rope'd or plain) (reused)
  float* kr   = alloc((size_t)1024 * 64);     // MQA k after rope
  float* sim  = alloc((size_t)16 * 1024 * 1024); // attention scores (reused)
  float* att  = alloc(M1);                    // merged attn out [*,INNER]
  float* gbuf = alloc((size_t)1024 * 4096);   // gated / gelu hidden (reused)
  float* lat  = alloc((size_t)256 * 1024);    // perceiver latents
  float* xp   = alloc(M1);                    // embeds + media_pos
  float* lm   = alloc((size_t)256 * 1024);    // latent LN scratch
  float* kvin = alloc((size_t)1280 * 1024);   // concat([xm,lm]) per t

  // ------------------------------ prologue --------------------------------
  gather_emb_kernel<<<NSEQ, 256, 0, stream>>>(token_emb, text, x, D);
  cumsum_media_kernel<<<1, 1024, 0, stream>>>(text, ttim, NSEQ);

  // ----------------------------- perceiver --------------------------------
  {
    int cnt = T * NPATCH * D;  // 1,048,576
    media_add_kernel<<<(cnt + 255) / 256, 256, 0, stream>>>(embeds, media_pos, xp, cnt);
    cnt = T * N_LAT * D;       // 262,144
    lat_init_kernel<<<(cnt + 255) / 256, 256, 0, stream>>>(latents, lat, cnt);
  }
  for (int l = 0; l < P_DEPTH; ++l) {
    const PercL& p = pl[l];
    layernorm_kernel<<<T * NPATCH, 256, 0, stream>>>(xp, p.nm_g, p.nm_b, xn, D);   // xm
    layernorm_kernel<<<T * N_LAT, 256, 0, stream>>>(lat, p.nl_g, p.nl_b, lm, D);   // lm
    gemm0(stream, lm, p.wq, qr, 256, INNER, D, D, INNER, INNER, 1.f, nullptr, 0, false);
    { int cnt = T * 320 * D;
      kvin_kernel<<<(cnt + 255) / 256, 256, 0, stream>>>(xn, lm, kvin, cnt); }
    gemm0(stream, kvin, p.wkv, proj, 1280, 2 * INNER, D, D, 2 * INNER, 2 * INNER,
          1.f, nullptr, 0, false);
    // sim[t,h,64,320] = SCALE * q @ k^T   (batch z1=t, z2=h)
    gemm(stream, qr, proj, sim, N_LAT, 320, DH, INNER, 2 * INNER, 320,
         (long long)N_LAT * INNER, DH, (long long)320 * 2 * INNER, DH,
         (long long)H * N_LAT * 320, (long long)N_LAT * 320,
         T, H, SCALE, nullptr, 0, true);
    softmax_kernel<<<T * H * N_LAT, 256, 0, stream>>>(sim, 320, N_LAT, 0, nullptr, 0);
    // out[t,i,h,d] -> att[(t*64+i), h*64+d]
    gemm(stream, sim, proj + INNER, att, N_LAT, DH, 320, 320, 2 * INNER, INNER,
         (long long)H * N_LAT * 320, (long long)N_LAT * 320,
         (long long)320 * 2 * INNER, DH,
         (long long)N_LAT * INNER, DH,
         T, H, 1.f, nullptr, 0, false);
    gemm0(stream, att, p.wo, lat, 256, D, INNER, INNER, D, D, 1.f, nullptr, 1, false);
    layernorm_kernel<<<T * N_LAT, 256, 0, stream>>>(lat, p.ff_g, p.ff_b, lm, D);
    gemm0(stream, lm, p.ff_w1, gbuf, 256, FF_IN, D, D, FF_IN, FF_IN, 1.f, nullptr, 0, false);
    gelu_kernel<<<(256 * FF_IN + 255) / 256, 256, 0, stream>>>(gbuf, 256 * FF_IN);
    gemm0(stream, gbuf, p.ff_w2, lat, 256, D, FF_IN, FF_IN, D, D, 1.f, nullptr, 1, false);
  }
  layernorm_kernel<<<T * N_LAT, 256, 0, stream>>>(lat, norm_g, norm_b, med, D);

  // --------------------------- transformer stack --------------------------
  for (int l = 0; l < DEPTH; ++l) {
    // ---- PaLM parallel block (residual accumulated directly into x) ----
    layernorm_kernel<<<NSEQ, 256, 0, stream>>>(x, ptb[l].g, nullptr, xn, D);
    gemm0(stream, xn, ptb[l].fused, proj, NSEQ, FUSED_W, D, D, FUSED_W, FUSED_W,
          1.f, nullptr, 0, false);
    rope_kernel<<<dim3(NSEQ, H), 64, 0, stream>>>(proj, qr, NSEQ, FUSED_W, 0);
    rope_kernel<<<dim3(NSEQ, 1), 64, 0, stream>>>(proj, kr, NSEQ, FUSED_W, K_OFF);
    // sim[h,n,j] = SCALE * q_h @ k^T  (k shared across heads: strideB=0)
    gemm(stream, qr, kr, sim, NSEQ, NSEQ, DH, DH, DH, NSEQ,
         (long long)NSEQ * DH, 0, 0, 0, (long long)NSEQ * NSEQ, 0,
         H, 1, SCALE, nullptr, 0, true);
    softmax_kernel<<<H * NSEQ, 256, 0, stream>>>(sim, NSEQ, NSEQ, 1, nullptr, 0);
    // out[h,n,d] -> att[n, h*64+d];  v lives inside proj (ldb = FUSED_W)
    gemm(stream, sim, proj + V_OFF, att, NSEQ, DH, NSEQ, NSEQ, FUSED_W, INNER,
         (long long)NSEQ * NSEQ, 0, 0, 0, (long long)DH, 0,
         H, 1, 1.f, nullptr, 0, false);
    gate_silu_kernel<<<(NSEQ * FF_IN + 255) / 256, 256, 0, stream>>>(
        proj, gbuf, NSEQ * FF_IN);
    gemm0(stream, att, ptb[l].attn_out, x, NSEQ, D, INNER, INNER, D, D,
          1.f, nullptr, 1, false);
    gemm0(stream, gbuf, ptb[l].ff_out, x, NSEQ, D, FF_IN, FF_IN, D, D,
          1.f, nullptr, 1, false);

    // ---- gated cross-attention ----
    if (hasG[l]) {
      const GcaP& g = gca[l];
      layernorm_kernel<<<NSEQ, 256, 0, stream>>>(x, g.a_g, g.a_b, xn, D);
      gemm0(stream, xn, g.wq, qr, NSEQ, INNER, D, D, INNER, INNER,
            1.f, nullptr, 0, false);
      gemm0(stream, med, g.wkv, proj, 256, 2 * INNER, D, D, 2 * INNER, 2 * INNER,
            1.f, nullptr, 0, false);
      // sim[h,n,j] = SCALE * q[:,h*64:(h+1)*64] @ k_h^T  (strided head views)
      gemm(stream, qr, proj, sim, NSEQ, 256, DH, INNER, 2 * INNER, 256,
           DH, 0, DH, 0, (long long)NSEQ * 256, 0,
           H, 1, SCALE, nullptr, 0, true);
      softmax_kernel<<<H * NSEQ, 256, 0, stream>>>(sim, 256, NSEQ, 2, ttim, N_LAT);
      gemm(stream, sim, proj + INNER, att, NSEQ, DH, 256, 256, 2 * INNER, INNER,
           (long long)NSEQ * 256, 0, DH, 0, (long long)DH, 0,
           H, 1, 1.f, nullptr, 0, false);
      // x += tanh(attn_gate) * (att @ wo)
      gemm0(stream, att, g.wo, x, NSEQ, D, INNER, INNER, D, D,
            1.f, g.attn_gate, 1, false);
      layernorm_kernel<<<NSEQ, 256, 0, stream>>>(x, g.ff_g, g.ff_b, xn, D);
      gemm0(stream, xn, g.ff_w1, gbuf, NSEQ, FF_IN, D, D, FF_IN, FF_IN,
            1.f, nullptr, 0, false);
      gelu_kernel<<<(NSEQ * FF_IN + 255) / 256, 256, 0, stream>>>(gbuf, NSEQ * FF_IN);
      // x += tanh(ff_gate) * (gelu(h) @ ff_w2)
      gemm0(stream, gbuf, g.ff_w2, x, NSEQ, D, FF_IN, FF_IN, D, D,
            1.f, g.ff_gate, 1, false);
    }
  }

  // ------------------------------ logits ----------------------------------
  layernorm_kernel<<<NSEQ, 256, 0, stream>>>(x, final_g, nullptr, xn, D);
  // logits = xf @ token_emb^T : [1024, 32000]
  gemm0(stream, xn, token_emb, (float*)d_out, NSEQ, V, D, D, D, V,
        1.f, nullptr, 0, true);
}
